// BallCritic_88673894793691
// MI455X (gfx1250) — matrix operand order repaired and checked by reading, not compiled, not run on hardware
//
#include <hip/hip_runtime.h>
#include <hip/hip_bf16.h>

// ---------------------------------------------------------------------------
// BallCritic on MI455X (gfx1250, wave32, WMMA f32_16x16x32_f16)
//
// Algebraic restructuring vs reference:
//   ce     = (tanh(emb) @ em_w + em_b)[cat]            -> 3x64 lookup table
//   msg_l1 = h_i @ (W1a - W1b) + h_j @ W1b + b1        -> per-node u,v GEMMs
// All matmuls run on v_wmma_f32_16x16x32_f16 (f16 in, f32 accum); operands are
// tanh-bounded so f16 is numerically safe.  tanh is branch-free (native
// v_tanh_f32 when available, else exp/rcp rational).
// ---------------------------------------------------------------------------

#define BS   1024
#define NN   64
#define HID  128
#define EMB  64
#define KNN  8
#define NC   3

typedef __attribute__((ext_vector_type(16))) _Float16 v16h;
typedef __attribute__((ext_vector_type(8)))  _Float16 v8h;
typedef __attribute__((ext_vector_type(4)))  _Float16 v4h;
typedef __attribute__((ext_vector_type(8)))  float    v8f;

// Branch-free tanh: prefer native gfx1250 V_TANH_F32, else exp/rcp rational.
__device__ __forceinline__ float fast_tanh(float x) {
#if __has_builtin(__builtin_amdgcn_tanhf)
  return __builtin_amdgcn_tanhf(x);
#elif __has_builtin(__builtin_amdgcn_tanh_f32)
  return __builtin_amdgcn_tanh_f32(x);
#else
  float xc = fminf(fmaxf(x, -10.0f), 10.0f);
  float e  = __expf(2.0f * xc);          // v_exp_f32 based, branch-free
#if __has_builtin(__builtin_amdgcn_rcpf)
  return (e - 1.0f) * __builtin_amdgcn_rcpf(e + 1.0f);
#else
  return (e - 1.0f) / (e + 1.0f);
#endif
#endif
}

__device__ __forceinline__ v8f vzero8() {
  v8f z;
#pragma unroll
  for (int r = 0; r < 8; ++r) z[r] = 0.0f;
  return z;
}

// A fragment: 16x32 f16, row-major source with leading dim ld (halves).
// ISA layout: lane<16 -> row=lane, K chunks [0..7],[16..23]; lane>=16 -> K [8..15],[24..31]
__device__ __forceinline__ v16h load_a_frag(const _Float16* base, int ld) {
  int lane = threadIdx.x & 31;
  int row  = lane & 15;
  int kc   = (lane >> 4) << 3;            // 0 or 8
  const _Float16* p = base + row * ld + kc;
  v8h lo = *(const v8h*)(p);              // K = kc .. kc+7
  v8h hi = *(const v8h*)(p + 16);         // K = kc+16 .. kc+23
  v16h r;
#pragma unroll
  for (int i = 0; i < 8; ++i) { r[i] = lo[i]; r[8 + i] = hi[i]; }
  return r;
}

// B fragment: 32x16 f16 from transposed weights Bt[N][K] (K contiguous).
// ISA layout: lane<16 -> col=lane, K chunks [0..7],[16..23]; lane>=16 -> K [8..15],[24..31]
__device__ __forceinline__ v16h load_b_frag(const _Float16* base, int ld) {
  int lane = threadIdx.x & 31;
  int col  = lane & 15;
  int kc   = (lane >> 4) << 3;            // 0 or 8
  const _Float16* p = base + col * ld + kc;
  v8h lo = *(const v8h*)(p);
  v8h hi = *(const v8h*)(p + 16);
  v16h r;
#pragma unroll
  for (int i = 0; i < 8; ++i) { r[i] = lo[i]; r[8 + i] = hi[i]; }
  return r;
}

__device__ __forceinline__ v8f wmma16(v16h a, v16h b, v8f c) {
  return __builtin_amdgcn_wmma_f32_16x16x32_f16(false, a, false, b, (short)0, c,
                                                false, false);
}

// ---------------------------------------------------------------------------
// kNN: top-8 nearest neighbours per node (matches jax top_k tie-breaking:
// strict < insertion keeps the lowest index first among equal distances).
// ---------------------------------------------------------------------------
__global__ void knn_kernel(const float* __restrict__ state, int* __restrict__ idx) {
  __shared__ float px[NN], py[NN];
  int b = blockIdx.x, i = threadIdx.x;
  const float* st = state + (size_t)b * NN * 3;
  px[i] = st[i * 3 + 0];
  py[i] = st[i * 3 + 1];
  __syncthreads();
  float bd[KNN];
  int   bi[KNN];
#pragma unroll
  for (int k = 0; k < KNN; ++k) { bd[k] = 3.0e38f; bi[k] = 0; }
  float xi = px[i], yi = py[i];
  for (int j = 0; j < NN; ++j) {
    float dx = xi - px[j], dy = yi - py[j];
    float d = dx * dx + dy * dy;
    if (j == i) d += 1.0e10f;
    if (d < bd[KNN - 1]) {
      int p = KNN - 1;
      while (p > 0 && d < bd[p - 1]) { bd[p] = bd[p - 1]; bi[p] = bi[p - 1]; --p; }
      bd[p] = d; bi[p] = j;
    }
  }
  int* o = idx + ((size_t)b * NN + i) * KNN;
#pragma unroll
  for (int k = 0; k < KNN; ++k) o[k] = bi[k];
}

// ---------------------------------------------------------------------------
// Per-branch weight prep: f16 transposed copies (so B-fragments have
// contiguous K), W1 factorization A1=W1a-W1b, B1=W1b, and ce3 lookup table.
// ---------------------------------------------------------------------------
__global__ void prep_kernel(const float* __restrict__ sp_w2,
                            const float* __restrict__ m_w1,
                            const float* __restrict__ m_w2,
                            const float* __restrict__ t_w1,
                            const float* __restrict__ emb,
                            const float* __restrict__ em_w,
                            const float* __restrict__ em_b,
                            _Float16* __restrict__ W2sp, _Float16* __restrict__ A1t,
                            _Float16* __restrict__ B1t,  _Float16* __restrict__ W2m,
                            _Float16* __restrict__ Wt1,  float* __restrict__ ce3) {
  int id = blockIdx.x * 256 + threadIdx.x;
  if (id < 16384) {
    int n = id >> 7, k = id & 127;
    W2sp[n * 128 + k] = (_Float16)sp_w2[k * 128 + n];
  } else if (id < 40960) {
    int e = id - 16384, n = e / 192, k = e % 192;
    A1t[n * 192 + k] = (_Float16)(m_w1[k * 128 + n] - m_w1[(k + 192) * 128 + n]);
  } else if (id < 65536) {
    int e = id - 40960, n = e / 192, k = e % 192;
    B1t[n * 192 + k] = (_Float16)m_w1[(k + 192) * 128 + n];
  } else if (id < 81920) {
    int e = id - 65536, n = e >> 7, k = e & 127;
    W2m[n * 128 + k] = (_Float16)m_w2[k * 128 + n];
  } else if (id < 98304) {
    int e = id - 81920, n = e >> 7, k = e & 127;
    Wt1[n * 128 + k] = (_Float16)t_w1[k * 128 + n];
  } else if (id < 98496) {
    int e = id - 98304, c = e >> 6, j = e & 63;
    float acc = em_b[j];
    for (int t = 0; t < 64; ++t) acc += fast_tanh(emb[c * 64 + t]) * em_w[t * 64 + j];
    ce3[e] = acc;
  }
}

// ---------------------------------------------------------------------------
// h kernel: layer1 (K=6, scalar FMA) -> tanh -> WMMA layer2 -> tanh -> h[:,0:128]
//           ce3 lookup -> tanh -> h[:,128:192].  1 block = 1 batch, 4 waves.
// ---------------------------------------------------------------------------
__global__ void h_kernel(const float* __restrict__ state,
                         const float* __restrict__ action,
                         const float* __restrict__ tars,
                         const float* __restrict__ sp_w1,
                         const float* __restrict__ sp_b1,
                         const float* __restrict__ sp_b2,
                         const float* __restrict__ ce3,
                         const _Float16* __restrict__ W2sp_t,
                         _Float16* __restrict__ h) {
  __shared__ __align__(16) _Float16 s1[NN * 128];   // tanh(layer1) 16KB
  __shared__ float s_in[NN * 6];
  __shared__ int   s_cat[NN];
  int b = blockIdx.x, tid = threadIdx.x;
  if (tid < NN) {
    int i = tid;
    const float* st = state + ((size_t)b * NN + i) * 3;
    s_in[i * 6 + 0] = st[0];
    s_in[i * 6 + 1] = st[1];
    s_cat[i]        = (int)st[2];
    const float* ac = action + (size_t)b * NN * 2 + i * 2;
    s_in[i * 6 + 2] = ac[0];
    s_in[i * 6 + 3] = ac[1];
    const float* ts = tars + ((size_t)b * NN + i) * 2;
    s_in[i * 6 + 4] = fast_tanh(ts[0]);
    s_in[i * 6 + 5] = fast_tanh(ts[1]);
  }
  __syncthreads();
  {
    int c = tid;                                    // one column per thread
    float w[6];
#pragma unroll
    for (int e = 0; e < 6; ++e) w[e] = sp_w1[e * 128 + c];
    float bc = sp_b1[c];
    for (int i = 0; i < NN; ++i) {
      float acc = bc;
#pragma unroll
      for (int e = 0; e < 6; ++e) acc += w[e] * s_in[i * 6 + e];
      s1[i * 128 + c] = (_Float16)fast_tanh(acc);
    }
  }
  __syncthreads();
  int wid = tid >> 5, lane = tid & 31;
  _Float16* hb = h + (size_t)b * NN * 192;
  for (int nt = 0; nt < 8; ++nt) {
    v8f acc = vzero8();
#pragma unroll
    for (int ks = 0; ks < 4; ++ks)
      acc = wmma16(load_a_frag(s1 + wid * 16 * 128 + ks * 32, 128),
                   load_b_frag(W2sp_t + nt * 16 * 128 + ks * 32, 128), acc);
    int col  = nt * 16 + (lane & 15);
    float b2 = sp_b2[col];
    int rb   = wid * 16 + ((lane >> 4) << 3);
#pragma unroll
    for (int r = 0; r < 8; ++r)
      hb[(rb + r) * 192 + col] = (_Float16)fast_tanh(acc[r] + b2);
  }
  // ce part, cols 128..191, packed 4-wide stores
  for (int e = tid; e < NN * 16; e += 128) {
    int i = e >> 4, j4 = (e & 15) * 4;
    const float* c3 = ce3 + s_cat[i] * 64 + j4;
    v4h t;
#pragma unroll
    for (int c = 0; c < 4; ++c) t[c] = (_Float16)fast_tanh(c3[c]);
    *(v4h*)(hb + i * 192 + 128 + j4) = t;
  }
}

// ---------------------------------------------------------------------------
// u/v kernel: u = h@A1 (+b1 folded at store), v = h@B1.  K=192 -> 6 k-steps.
// ---------------------------------------------------------------------------
__global__ void uv_kernel(const _Float16* __restrict__ h,
                          const _Float16* __restrict__ A1t,
                          const _Float16* __restrict__ B1t,
                          const float* __restrict__ m_b1,
                          _Float16* __restrict__ u, _Float16* __restrict__ v) {
  __shared__ __align__(16) _Float16 s_h[NN * 192];  // 24KB
  int b = blockIdx.x, tid = threadIdx.x;
  {
    const uint4* src = (const uint4*)(h + (size_t)b * NN * 192);
    uint4* dst = (uint4*)s_h;
    for (int e = tid; e < NN * 192 / 8; e += 128) dst[e] = src[e];
  }
  __syncthreads();
  int wid = tid >> 5, lane = tid & 31;
  size_t base = (size_t)b * NN * 128;
  for (int nt = 0; nt < 8; ++nt) {
    v8f au = vzero8(), av = vzero8();
#pragma unroll
    for (int ks = 0; ks < 6; ++ks) {
      v16h a = load_a_frag(s_h + wid * 16 * 192 + ks * 32, 192);
      au = wmma16(a, load_b_frag(A1t + nt * 16 * 192 + ks * 32, 192), au);
      av = wmma16(a, load_b_frag(B1t + nt * 16 * 192 + ks * 32, 192), av);
    }
    int col  = nt * 16 + (lane & 15);
    float b1 = m_b1[col];
    int rb   = wid * 16 + ((lane >> 4) << 3);
#pragma unroll
    for (int r = 0; r < 8; ++r) {
      u[base + (rb + r) * 128 + col] = (_Float16)(au[r] + b1);
      v[base + (rb + r) * 128 + col] = (_Float16)av[r];
    }
  }
}

// ---------------------------------------------------------------------------
// Edge kernel: per 16-edge M-tile (exactly 2 nodes): gather u_i+v_j, tanh,
// WMMA against m_w2, per-lane max over the lane's 8 rows (one node), tanh.
// ---------------------------------------------------------------------------
__global__ void msg_kernel(const _Float16* __restrict__ u,
                           const _Float16* __restrict__ v,
                           const int* __restrict__ idx,
                           const _Float16* __restrict__ W2m_t,
                           const float* __restrict__ m_b2,
                           _Float16* __restrict__ x) {
  __shared__ __align__(16) _Float16 s_w[128 * 128];      // 32KB
  __shared__ __align__(16) _Float16 s_a[4][16 * 128];    // 16KB (per wave)
  int b = blockIdx.x, tid = threadIdx.x, wid = tid >> 5, lane = tid & 31;
  {
    const uint4* src = (const uint4*)W2m_t;
    uint4* dst = (uint4*)s_w;
    for (int e = tid; e < 128 * 128 / 8; e += 128) dst[e] = src[e];
  }
  __syncthreads();
  const _Float16* ub = u + (size_t)b * NN * 128;
  const _Float16* vb = v + (size_t)b * NN * 128;
  const int* ib = idx + (size_t)b * NN * KNN;
  _Float16* xb = x + (size_t)b * NN * 128;
  for (int g = wid; g < NN * KNN / 16; g += 4) {         // 8 tiles per wave
    int c0 = lane * 4;
#pragma unroll
    for (int m = 0; m < 16; ++m) {
      int E = g * 16 + m;
      int i = E >> 3, k = E & 7;
      int j = ib[i * KNN + k];
      v4h uu = *(const v4h*)(ub + i * 128 + c0);
      v4h vv = *(const v4h*)(vb + j * 128 + c0);
      v4h t;
#pragma unroll
      for (int c = 0; c < 4; ++c)
        t[c] = (_Float16)fast_tanh((float)uu[c] + (float)vv[c]);
      *(v4h*)(&s_a[wid][m * 128 + c0]) = t;
    }
    __syncthreads();
    for (int nt = 0; nt < 8; ++nt) {
      v8f acc = vzero8();
#pragma unroll
      for (int ks = 0; ks < 4; ++ks)
        acc = wmma16(load_a_frag(&s_a[wid][ks * 32], 128),
                     load_b_frag(s_w + nt * 16 * 128 + ks * 32, 128), acc);
      float mv = acc[0];
#pragma unroll
      for (int r = 1; r < 8; ++r) mv = fmaxf(mv, acc[r]);
      int col  = nt * 16 + (lane & 15);
      int node = g * 2 + (lane >> 4);                    // lane-half <-> node
      xb[node * 128 + col] = (_Float16)fast_tanh(mv + m_b2[col]);
    }
    __syncthreads();
  }
}

// ---------------------------------------------------------------------------
// Head: tanh(x@t_w1+b1) dot t_w2 + b2.  WMMA then cross-lane shuffle dot.
// ---------------------------------------------------------------------------
__global__ void tail_kernel(const _Float16* __restrict__ x,
                            const _Float16* __restrict__ Wt1_t,
                            const float* __restrict__ t_b1,
                            const float* __restrict__ t_w2,
                            const float* __restrict__ t_b2,
                            float* __restrict__ out) {
  __shared__ __align__(16) _Float16 s_w[128 * 128];  // 32KB
  __shared__ __align__(16) _Float16 s_x[NN * 128];   // 16KB
  int b = blockIdx.x, tid = threadIdx.x, wid = tid >> 5, lane = tid & 31;
  {
    const uint4* ws = (const uint4*)Wt1_t;
    uint4* wd = (uint4*)s_w;
    for (int e = tid; e < 128 * 128 / 8; e += 128) wd[e] = ws[e];
    const uint4* xs = (const uint4*)(x + (size_t)b * NN * 128);
    uint4* xd = (uint4*)s_x;
    for (int e = tid; e < NN * 128 / 8; e += 128) xd[e] = xs[e];
  }
  __syncthreads();
  float racc[8];
#pragma unroll
  for (int r = 0; r < 8; ++r) racc[r] = 0.0f;
  for (int nt = 0; nt < 8; ++nt) {
    v8f acc = vzero8();
#pragma unroll
    for (int ks = 0; ks < 4; ++ks)
      acc = wmma16(load_a_frag(s_x + wid * 16 * 128 + ks * 32, 128),
                   load_b_frag(s_w + nt * 16 * 128 + ks * 32, 128), acc);
    int col  = nt * 16 + (lane & 15);
    float b1 = t_b1[col], w2 = t_w2[col];
#pragma unroll
    for (int r = 0; r < 8; ++r) racc[r] += fast_tanh(acc[r] + b1) * w2;
  }
  // reduce the 16-lane column partials within each lane-half (wave32 shuffles)
#pragma unroll
  for (int off = 1; off < 16; off <<= 1)
#pragma unroll
    for (int r = 0; r < 8; ++r) racc[r] += __shfl_xor(racc[r], off, 32);
  if ((lane & 15) == 0) {
    int rb = wid * 16 + ((lane >> 4) << 3);
    float tb2 = t_b2[0];
#pragma unroll
    for (int r = 0; r < 8; ++r) out[(size_t)b * NN + rb + r] = racc[r] + tb2;
  }
}

// ---------------------------------------------------------------------------
// Launch.  Input order (setup_inputs insertion order):
//   0 state, 1 action, 2 tar_scores, then q1 then q2, each:
//   sp_w1, sp_b1, sp_w2, sp_b2, emb, em_w, em_b, m_w1, m_b1, m_w2, m_b2,
//   t_w1, t_b1, t_w2, t_b2
// Workspace use ~78 MB (idx 2MB, h 24MB, u/v/x 16MB each, weights <0.2MB).
// ---------------------------------------------------------------------------
extern "C" void kernel_launch(void* const* d_in, const int* in_sizes, int n_in,
                              void* d_out, int out_size, void* d_ws, size_t ws_size,
                              hipStream_t stream) {
  (void)in_sizes; (void)n_in; (void)out_size; (void)ws_size;
  const float* state  = (const float*)d_in[0];
  const float* action = (const float*)d_in[1];
  const float* tars   = (const float*)d_in[2];

  char* ws = (char*)d_ws;
  size_t off = 0;
  auto take = [&](size_t bytes) -> char* {
    char* p = ws + off;
    off = (off + bytes + 255) & ~(size_t)255;
    return p;
  };
  int*      idxb = (int*)     take((size_t)BS * NN * KNN * 4);
  _Float16* hbuf = (_Float16*)take((size_t)BS * NN * 192 * 2);
  _Float16* ubuf = (_Float16*)take((size_t)BS * NN * 128 * 2);
  _Float16* vbuf = (_Float16*)take((size_t)BS * NN * 128 * 2);
  _Float16* xbuf = (_Float16*)take((size_t)BS * NN * 128 * 2);
  _Float16* W2sp = (_Float16*)take(128 * 128 * 2);
  _Float16* A1t  = (_Float16*)take(128 * 192 * 2);
  _Float16* B1t  = (_Float16*)take(128 * 192 * 2);
  _Float16* W2m  = (_Float16*)take(128 * 128 * 2);
  _Float16* Wt1  = (_Float16*)take(128 * 128 * 2);
  float*    ce3  = (float*)   take(NC * EMB * 4);

  knn_kernel<<<BS, NN, 0, stream>>>(state, idxb);

  for (int br = 0; br < 2; ++br) {
    const float* const* P = (const float* const*)&d_in[3 + br * 15];
    const float* sp_w1 = P[0];  const float* sp_b1 = P[1];
    const float* sp_w2 = P[2];  const float* sp_b2 = P[3];
    const float* emb   = P[4];  const float* em_w  = P[5];
    const float* em_b  = P[6];  const float* m_w1  = P[7];
    const float* m_b1  = P[8];  const float* m_w2  = P[9];
    const float* m_b2  = P[10]; const float* t_w1  = P[11];
    const float* t_b1  = P[12]; const float* t_w2  = P[13];
    const float* t_b2  = P[14];

    prep_kernel<<<385, 256, 0, stream>>>(sp_w2, m_w1, m_w2, t_w1, emb, em_w, em_b,
                                         W2sp, A1t, B1t, W2m, Wt1, ce3);
    h_kernel<<<BS, 128, 0, stream>>>(state, action, tars, sp_w1, sp_b1, sp_b2,
                                     ce3, W2sp, hbuf);
    uv_kernel<<<BS, 128, 0, stream>>>(hbuf, A1t, B1t, m_b1, ubuf, vbuf);
    msg_kernel<<<BS, 128, 0, stream>>>(ubuf, vbuf, idxb, W2m, m_b2, xbuf);
    tail_kernel<<<BS, 128, 0, stream>>>(xbuf, Wt1, t_b1, t_w2, t_b2,
                                        (float*)d_out + (size_t)br * BS * NN);
  }
}
